// AttnBlock_7206955122865
// MI455X (gfx1250) — matrix operand order repaired and checked by reading, not compiled, not run on hardware
//
#include <hip/hip_runtime.h>
#include <hip/hip_bf16.h>
#include <math.h>
#include <cstdint>

// ---------- problem constants (reference: B=4, C=512, H=W=64) ----------
#define NB   4
#define C_   512
#define HW   4096       // H*W
#define NG   32         // groups
#define CG   16         // channels per group = C_/NG
#define EPSV 1e-6f

typedef __bf16 bf16t;
typedef __attribute__((ext_vector_type(16))) __bf16 v16bf;
typedef __attribute__((ext_vector_type(8)))  __bf16 v8bf;
typedef __attribute__((ext_vector_type(8)))  float  v8f;

__device__ __forceinline__ v8f zero8() {
    v8f z = {0.f,0.f,0.f,0.f,0.f,0.f,0.f,0.f};
    return z;
}

// ---- CDNA5 async global->LDS copy (ASYNCcnt-tracked DMA, bypasses VGPRs) ----
__device__ __forceinline__ void async_ld_b128(uint32_t lds_addr, const bf16t* g) {
    asm volatile("global_load_async_to_lds_b128 %0, %1, off"
                 :: "v"(lds_addr), "v"((unsigned long long)(uintptr_t)g)
                 : "memory");
}
__device__ __forceinline__ void wait_async0() {
    asm volatile("s_wait_asynccnt 0" ::: "memory");
}

// ---------------- f32 -> bf16 weight conversion ----------------
__global__ void cvt_f32_bf16_kernel(const float* __restrict__ src,
                                    bf16t* __restrict__ dst, int n) {
    int i = blockIdx.x * 256 + threadIdx.x;
    if (i < n) dst[i] = (bf16t)src[i];
}

// ---------------- GroupNorm: per-(b,g) mean / rsqrt(var) ----------------
__global__ void gn_stats_kernel(const float* __restrict__ x,
                                float2* __restrict__ stats) {
    const int g = blockIdx.x, b = blockIdx.y;
    const float* xp = x + ((size_t)b * C_ + (size_t)g * CG) * HW;  // contiguous CG*HW span
    const int total = CG * HW;
    float s = 0.f, ss = 0.f;
    for (int i = threadIdx.x; i < total; i += 256) {
        float v = xp[i];
        s += v; ss += v * v;
    }
    __shared__ float rs[256], rq[256];
    rs[threadIdx.x] = s; rq[threadIdx.x] = ss;
    __syncthreads();
    for (int off = 128; off > 0; off >>= 1) {
        if (threadIdx.x < off) {
            rs[threadIdx.x] += rs[threadIdx.x + off];
            rq[threadIdx.x] += rq[threadIdx.x + off];
        }
        __syncthreads();
    }
    if (threadIdx.x == 0) {
        float mean = rs[0] / (float)total;
        float var  = rq[0] / (float)total - mean * mean;
        float2 st; st.x = mean; st.y = rsqrtf(var + EPSV);
        stats[b * NG + g] = st;
    }
}

// ---------------- GroupNorm apply: h = gn(x) as bf16 [B,C,N] ----------------
__global__ void gn_apply_kernel(const float* __restrict__ x,
                                const float* __restrict__ gamma,
                                const float* __restrict__ beta,
                                const float2* __restrict__ stats,
                                bf16t* __restrict__ h) {
    size_t idx = (size_t)blockIdx.x * 256 + threadIdx.x;   // one float4 each
    size_t e = idx * 4;
    int c = (int)((e / HW) % C_);
    int b = (int)(e / ((size_t)C_ * HW));
    float2 st = stats[b * NG + c / CG];
    float ga = gamma[c] * st.y;
    float be = beta[c] - st.x * ga;
    const float4 xv = *(const float4*)(x + e);
    h[e + 0] = (bf16t)(xv.x * ga + be);
    h[e + 1] = (bf16t)(xv.y * ga + be);
    h[e + 2] = (bf16t)(xv.z * ga + be);
    h[e + 3] = (bf16t)(xv.w * ga + be);
}

// ---------------- row softmax: P = softmax(S row), bf16 out ----------------
__global__ void softmax_kernel(const float* __restrict__ S,
                               bf16t* __restrict__ P) {
    const int t = threadIdx.x;
    const float* s = S + (size_t)blockIdx.x * HW;
    bf16t*       p = P + (size_t)blockIdx.x * HW;
    float v[16];
    float m = -3.4e38f;
#pragma unroll
    for (int i = 0; i < 16; ++i) { v[i] = s[t + i * 256]; m = fmaxf(m, v[i]); }
    __shared__ float red[256];
    red[t] = m; __syncthreads();
    for (int off = 128; off > 0; off >>= 1) {
        if (t < off) red[t] = fmaxf(red[t], red[t + off]);
        __syncthreads();
    }
    m = red[0]; __syncthreads();
    float sum = 0.f;
#pragma unroll
    for (int i = 0; i < 16; ++i) { v[i] = __expf(v[i] - m); sum += v[i]; }
    red[t] = sum; __syncthreads();
    for (int off = 128; off > 0; off >>= 1) {
        if (t < off) red[t] += red[t + off];
        __syncthreads();
    }
    float inv = 1.0f / red[0];
#pragma unroll
    for (int i = 0; i < 16; ++i) p[t + i * 256] = (bf16t)(v[i] * inv);
}

// ---------------- generic bf16 WMMA GEMM ----------------
// D[m,n] = alpha * sum_k A[m,k]*B[k,n] (+ bias[m]) (+ resid[m,n])
//  A[m,k] = A[m*lda + k]   (A_COL=0)  |  A[k*lda + m] (A_COL=1)
//  B[k,n] = B[k*ldb + n]   (B_COL=0)  |  B[n*ldb + k] (B_COL=1)
// 128x128 tile per 256-thread block (8 wave32s in 4x2), K-step 32,
// double-buffered LDS fed by async global->LDS DMA for the contiguous operands.
// All per-thread staging addresses hoisted out of the k-loop; per-step offset
// is uniform (SALU) so the hot loop carries no vector 64-bit multiplies.
#define AS_LD 40     // 128 x 32 bf16 tile, padded (80B rows, 16B aligned frags)
#define BS_LD 136    // 32 x 128 bf16 tile, padded (272B rows)

template<bool A_COL, bool B_COL, bool HAS_BIAS, bool HAS_RESID, bool OUT_F32>
__global__ __launch_bounds__(256)
void gemm_bf16_kernel(const bf16t* __restrict__ A, const bf16t* __restrict__ B,
                      const float* __restrict__ bias, const float* __restrict__ resid,
                      float* __restrict__ outF, bf16t* __restrict__ outB,
                      int K, int lda, int ldb, int ldd, float alpha,
                      long aBatch, long bBatch, long dBatch) {
    const int tid  = threadIdx.x;
    const int lane = tid & 31;
    const int wid  = tid >> 5;
    const int wm   = wid >> 1;        // 0..3 : 32 rows each
    const int wn   = wid & 1;         // 0..1 : 64 cols each
    const int m0   = blockIdx.y * 128;
    const int n0   = blockIdx.x * 128;
    const int bz   = blockIdx.z;

    const bf16t* Ap = A + (size_t)bz * aBatch;
    const bf16t* Bp = B + (size_t)bz * bBatch;

    __shared__ __align__(16) bf16t As[2][128 * AS_LD];
    __shared__ __align__(16) bf16t Bs[2][32 * BS_LD];
    const uint32_t asBase = (uint32_t)(uintptr_t)&As[0][0];   // LDS byte address (bits[31:0])
    const uint32_t bsBase = (uint32_t)(uintptr_t)&Bs[0][0];
    const uint32_t asBuf  = 128 * AS_LD * 2;                  // bytes per buffer
    const uint32_t bsBuf  = 32 * BS_LD * 2;

    // ---- hoisted per-thread staging descriptors (2 chunks each operand) ----
    const bf16t* aGlb[2]; uint32_t aLdsOff[2]; int aIdx[2][2];   // aIdx: {mc,k} for col path
    const bf16t* bGlb[2]; uint32_t bLdsOff[2]; int bIdx[2][2];
    size_t aStep, bStep;                                         // elements per k-step
#pragma unroll
    for (int it = 0; it < 2; ++it) {
        int chunk = tid + it * 256;                              // 0..511
        if (!A_COL) {
            int m = chunk >> 2, kc = (chunk & 3) * 8;
            aGlb[it]    = Ap + (size_t)(m0 + m) * lda + kc;
            aLdsOff[it] = asBase + (uint32_t)(m * AS_LD + kc) * 2;
        } else {
            int k = chunk >> 4, mc = (chunk & 15) * 8;
            aGlb[it]   = Ap + (size_t)k * lda + m0 + mc;
            aIdx[it][0] = mc; aIdx[it][1] = k;
        }
        if (!B_COL) {
            int k = chunk >> 4, nc = (chunk & 15) * 8;
            bGlb[it]    = Bp + (size_t)k * ldb + n0 + nc;
            bLdsOff[it] = bsBase + (uint32_t)(k * BS_LD + nc) * 2;
        } else {
            int n = chunk >> 2, kc = (chunk & 3) * 8;
            bGlb[it]   = Bp + (size_t)(n0 + n) * ldb + kc;
            bIdx[it][0] = kc; bIdx[it][1] = n;
        }
    }
    aStep = A_COL ? (size_t)32 * lda : (size_t)32;
    bStep = B_COL ? (size_t)32 : (size_t)32 * ldb;   // col: +32 along k (contig); row: +32 rows

    auto stage = [&](int kt, int buf) {              // stage k-tile `kt` into buffer `buf`
        const size_t aOff = (size_t)kt * aStep;      // uniform -> SALU
        const size_t bOff = (size_t)kt * bStep;
        if (!A_COL) {
#pragma unroll
            for (int it = 0; it < 2; ++it)
                async_ld_b128(aLdsOff[it] + buf * asBuf, aGlb[it] + aOff);
        } else {
#pragma unroll
            for (int it = 0; it < 2; ++it) {
                v8bf val = *(const v8bf*)(aGlb[it] + aOff);
#pragma unroll
                for (int e = 0; e < 8; ++e)
                    As[buf][(aIdx[it][0] + e) * AS_LD + aIdx[it][1]] = val[e];
            }
        }
        if (!B_COL) {
#pragma unroll
            for (int it = 0; it < 2; ++it)
                async_ld_b128(bLdsOff[it] + buf * bsBuf, bGlb[it] + bOff);
        } else {
#pragma unroll
            for (int it = 0; it < 2; ++it) {
                v8bf val = *(const v8bf*)(bGlb[it] + bOff);
#pragma unroll
                for (int e = 0; e < 8; ++e)
                    Bs[buf][(bIdx[it][0] + e) * BS_LD + bIdx[it][1]] = val[e];
            }
        }
    };

    v8f acc[2][4];
#pragma unroll
    for (int i = 0; i < 2; ++i)
#pragma unroll
        for (int j = 0; j < 4; ++j) acc[i][j] = zero8();

    // ---- pipelined mainloop: issue kt+1 async, compute kt, wait, barrier ----
    const int KT = K / 32;
    stage(0, 0);
    wait_async0();
    __syncthreads();

#pragma unroll 2
    for (int kt = 0; kt < KT; ++kt) {
        const int cur = kt & 1;
        if (kt + 1 < KT)                 // overlap next tile's DMA with the WMMAs
            stage(kt + 1, cur ^ 1);

        // ---- fragments per ISA register layouts ----
        union F16 { v16bf v; v8bf h[2]; };
        F16 af[2], bfr[4];
        const bf16t* asP = &As[cur][0];
        const bf16t* bsP = &Bs[cur][0];
        const int ak0  = (lane & 16) ? 8 : 0;     // A: lanes<16 K{0..7,16..23}, else {8..15,24..31}
        const int arow = wm * 32 + (lane & 15);
#pragma unroll
        for (int tm = 0; tm < 2; ++tm) {
            af[tm].h[0] = *(const v8bf*)&asP[(arow + tm * 16) * AS_LD + ak0];
            af[tm].h[1] = *(const v8bf*)&asP[(arow + tm * 16) * AS_LD + ak0 + 16];
        }
#pragma unroll
        for (int tn = 0; tn < 4; ++tn) {          // B: lane l = col N=l%16, K same order as A
            int bn = wn * 64 + tn * 16 + (lane & 15);
#pragma unroll
            for (int s = 0; s < 8; ++s) {
                bfr[tn].h[0][s] = bsP[(ak0 + s) * BS_LD + bn];
                bfr[tn].h[1][s] = bsP[(ak0 + 16 + s) * BS_LD + bn];
            }
        }
#pragma unroll
        for (int tm = 0; tm < 2; ++tm)
#pragma unroll
            for (int tn = 0; tn < 4; ++tn)
                acc[tm][tn] = __builtin_amdgcn_wmma_f32_16x16x32_bf16(
                    false, af[tm].v, false, bfr[tn].v,
                    (short)0, acc[tm][tn], false, false);

        wait_async0();      // own async loads for kt+1 landed in LDS
        __syncthreads();    // everyone's landed; everyone done reading buf cur
    }

    // ---- epilogue: C/D layout = VGPR r -> M=r(+8 for lanes>=16), N=lane%16 ----
    const int mlo  = (lane & 16) ? 8 : 0;
    const int ncol = lane & 15;
#pragma unroll
    for (int tm = 0; tm < 2; ++tm)
#pragma unroll
        for (int tn = 0; tn < 4; ++tn)
#pragma unroll
            for (int r = 0; r < 8; ++r) {
                int mm = m0 + wm * 32 + tm * 16 + mlo + r;
                int nn = n0 + wn * 64 + tn * 16 + ncol;
                float val = alpha * acc[tm][tn][r];
                if (HAS_BIAS) val += bias[mm];
                size_t di = (size_t)bz * dBatch + (size_t)mm * ldd + nn;
                if (HAS_RESID) val += resid[di];
                if (OUT_F32) outF[di] = val;
                else         outB[di] = (bf16t)val;
            }
}

// ---------------- host orchestration ----------------
extern "C" void kernel_launch(void* const* d_in, const int* in_sizes, int n_in,
                              void* d_out, int out_size, void* d_ws, size_t ws_size,
                              hipStream_t stream) {
    (void)in_sizes; (void)n_in; (void)out_size; (void)ws_size;
    const float* x     = (const float*)d_in[0];
    const float* gamma = (const float*)d_in[1];
    const float* beta  = (const float*)d_in[2];
    const float* wq = (const float*)d_in[3];
    const float* bq = (const float*)d_in[4];
    const float* wk = (const float*)d_in[5];
    const float* bk = (const float*)d_in[6];
    const float* wv = (const float*)d_in[7];
    const float* bv = (const float*)d_in[8];
    const float* wo = (const float*)d_in[9];
    const float* bo = (const float*)d_in[10];
    float* out = (float*)d_out;

    char* ws = (char*)d_ws;
    size_t off = 0;
    auto wsalloc = [&](size_t bytes) -> void* {
        size_t cur = off;
        off = (off + bytes + 255) & ~(size_t)255;
        return (void*)(ws + cur);
    };

    const size_t CN = (size_t)C_ * HW;
    bf16t* wq_b = (bf16t*)wsalloc((size_t)C_ * C_ * 2);
    bf16t* wk_b = (bf16t*)wsalloc((size_t)C_ * C_ * 2);
    bf16t* wv_b = (bf16t*)wsalloc((size_t)C_ * C_ * 2);
    bf16t* wo_b = (bf16t*)wsalloc((size_t)C_ * C_ * 2);
    bf16t* h_b  = (bf16t*)wsalloc((size_t)NB * CN * 2);
    bf16t* q_b  = (bf16t*)wsalloc((size_t)NB * CN * 2);
    bf16t* k_b  = (bf16t*)wsalloc((size_t)NB * CN * 2);
    bf16t* v_b  = (bf16t*)wsalloc((size_t)NB * CN * 2);
    float* S    = (float*)wsalloc((size_t)HW * HW * 4);   // 64 MB, L2-resident per batch
    bf16t* P    = (bf16t*)wsalloc((size_t)HW * HW * 2);   // 32 MB
    bf16t* ao   = (bf16t*)wsalloc(CN * 2);                // per-batch attention output
    float2* stats = (float2*)wsalloc((size_t)NB * NG * sizeof(float2));

    // weights -> bf16
    const int wcnt = C_ * C_;
    cvt_f32_bf16_kernel<<<(wcnt + 255) / 256, 256, 0, stream>>>(wq, wq_b, wcnt);
    cvt_f32_bf16_kernel<<<(wcnt + 255) / 256, 256, 0, stream>>>(wk, wk_b, wcnt);
    cvt_f32_bf16_kernel<<<(wcnt + 255) / 256, 256, 0, stream>>>(wv, wv_b, wcnt);
    cvt_f32_bf16_kernel<<<(wcnt + 255) / 256, 256, 0, stream>>>(wo, wo_b, wcnt);

    // group norm
    gn_stats_kernel<<<dim3(NG, NB), 256, 0, stream>>>(x, stats);
    const int total4 = (int)((size_t)NB * CN / 4);
    gn_apply_kernel<<<total4 / 256, 256, 0, stream>>>(x, gamma, beta, stats, h_b);

    // q/k/v 1x1 convs: D[d,n] = W[d,:] . H[:,n] + b[d]  (M=C, N=HW, K=C, batched over grid.z)
    dim3 gqkv(HW / 128, C_ / 128, NB);
    gemm_bf16_kernel<false,false,true,false,false><<<gqkv, 256, 0, stream>>>(
        wq_b, h_b, bq, nullptr, nullptr, q_b, C_, C_, HW, HW, 1.0f, 0L, (long)CN, (long)CN);
    gemm_bf16_kernel<false,false,true,false,false><<<gqkv, 256, 0, stream>>>(
        wk_b, h_b, bk, nullptr, nullptr, k_b, C_, C_, HW, HW, 1.0f, 0L, (long)CN, (long)CN);
    gemm_bf16_kernel<false,false,true,false,false><<<gqkv, 256, 0, stream>>>(
        wv_b, h_b, bv, nullptr, nullptr, v_b, C_, C_, HW, HW, 1.0f, 0L, (long)CN, (long)CN);

    const float scale = 1.0f / sqrtf((float)C_);

    // attention, serialized per batch so S (64MB) and P (32MB) stay L2-resident
    for (int b = 0; b < NB; ++b) {
        const bf16t* qb = q_b + (size_t)b * CN;
        const bf16t* kb = k_b + (size_t)b * CN;
        const bf16t* vb = v_b + (size_t)b * CN;

        // S[i,j] = scale * sum_c q[c,i] k[c,j]  (A col-major, M=N=HW, K=C)
        gemm_bf16_kernel<true,false,false,false,true><<<dim3(HW / 128, HW / 128, 1), 256, 0, stream>>>(
            qb, kb, nullptr, nullptr, S, nullptr, C_, HW, HW, HW, scale, 0L, 0L, 0L);

        // P = softmax rows of S
        softmax_kernel<<<HW, 256, 0, stream>>>(S, P);

        // ao[c,i] = sum_j v[c,j] P[i,j]  (B col-major: B[j,i] = P[i*HW + j]; M=C, N=HW, K=HW)
        gemm_bf16_kernel<false,true,false,false,false><<<dim3(HW / 128, C_ / 128, 1), 256, 0, stream>>>(
            vb, P, nullptr, nullptr, nullptr, ao, HW, HW, HW, HW, 1.0f, 0L, 0L, 0L);

        // out[d,n] = x[d,n] + wo[d,:] . ao[:,n] + bo[d]
        gemm_bf16_kernel<false,false,true,true,true><<<dim3(HW / 128, C_ / 128, 1), 256, 0, stream>>>(
            wo_b, ao, bo, x + (size_t)b * CN, out + (size_t)b * CN, nullptr,
            C_, C_, HW, HW, 1.0f, 0L, 0L, 0L);
    }
}
